// ModuleDiGraph_27419071218542
// MI455X (gfx1250) — matrix operand order repaired and checked by reading, not compiled, not run on hardware
//
#include <hip/hip_runtime.h>

// ---------------------------------------------------------------------------
// CDNA5 / gfx1250 implementation of MLP + ResGatedGraphConv + scorer.
//  - weights pre-packed to bf16 once per call (prep kernel)
//  - dense node pipeline: v_wmma_f32_16x16x32_bf16, f32 accumulate
//  - 64x64 projection weights staged to LDS via GLOBAL_LOAD_ASYNC_TO_LDS_B128
//    (ASYNCcnt / s_wait_asynccnt) when the builtin is available
//  - edge phase: L2-resident float4 gathers + global_atomic_add_f32 scatter
// ---------------------------------------------------------------------------

typedef __attribute__((ext_vector_type(16))) __bf16 v16bf;
typedef __attribute__((ext_vector_type(8)))  float  v8f;
typedef int v4i __attribute__((ext_vector_type(4)));

#define F_IN 128
#define HDIM 64
#define W1_ELEMS (HDIM * F_IN)    /* 8192  */
#define WS_ELEMS (HDIM * HDIM)    /* 4096  */
#define PACK_ELEMS (W1_ELEMS + 5 * WS_ELEMS)  /* 28672 bf16 */

// ---- optional CDNA5 async global->LDS path --------------------------------
// Builtin prototype (from toolchain diagnostic): (v4i AS1*, v4i AS3*, Ii, Ii).
#if __has_builtin(__builtin_amdgcn_global_load_async_to_lds_b128)
#define HAVE_ASYNC_LDS 1
typedef __attribute__((address_space(1))) v4i* as1_v4i;
typedef __attribute__((address_space(3))) v4i* as3_v4i;
// generic->AS1: value-preserving 64-bit; generic->AS3: low 32 bits of a
// generic LDS address are the LDS byte offset (aperture rule), so truncate.
#define ASYNC_COPY_B128(g, l)                                    \
  __builtin_amdgcn_global_load_async_to_lds_b128(                \
      (as1_v4i)(unsigned long long)(g),                          \
      (as3_v4i)(unsigned)(unsigned long long)(l), 0, 0)
#else
#define HAVE_ASYNC_LDS 0
#endif

#if __has_builtin(__builtin_amdgcn_s_wait_asynccnt)
#define WAIT_ASYNC() __builtin_amdgcn_s_wait_asynccnt(0)
#else
#define WAIT_ASYNC() asm volatile("s_wait_asynccnt 0" ::: "memory")
#endif

__device__ __forceinline__ v8f wmma_bf16(v16bf a, v16bf b, v8f c) {
  // (neg_a, A, neg_b, B, c_mod, C, reuse_a, reuse_b)
  return __builtin_amdgcn_wmma_f32_16x16x32_bf16(false, a, false, b, (short)0, c,
                                                 false, false);
}

// A-fragment (16x32 bf16) from a row-major f32 matrix row pointer.
// Layout (ISA 7.12.2, 16-bit A 16x32): lane L holds row M = L%16,
// halves 0..7  -> K = kbase + (L>>4)*8 + {0..7}
// halves 8..15 -> K = kbase + 16 + (L>>4)*8 + {0..7}
__device__ __forceinline__ v16bf load_a_f32row(const float* __restrict__ row,
                                               int kbase, int kgrp) {
  const float* p = row + kbase + kgrp * 8;
  float4 f0 = *(const float4*)(p);
  float4 f1 = *(const float4*)(p + 4);
  float4 f2 = *(const float4*)(p + 16);
  float4 f3 = *(const float4*)(p + 20);
  v16bf a;
  a[0]  = (__bf16)f0.x; a[1]  = (__bf16)f0.y; a[2]  = (__bf16)f0.z; a[3]  = (__bf16)f0.w;
  a[4]  = (__bf16)f1.x; a[5]  = (__bf16)f1.y; a[6]  = (__bf16)f1.z; a[7]  = (__bf16)f1.w;
  a[8]  = (__bf16)f2.x; a[9]  = (__bf16)f2.y; a[10] = (__bf16)f2.z; a[11] = (__bf16)f2.w;
  a[12] = (__bf16)f3.x; a[13] = (__bf16)f3.y; a[14] = (__bf16)f3.z; a[15] = (__bf16)f3.w;
  return a;
}

// A-fragment from a 16x64 bf16 row-major LDS tile (inter-layer staging).
__device__ __forceinline__ v16bf load_a_lds(const __bf16* __restrict__ h,
                                            int kbase, int lane) {
  const __bf16* p = h + (lane & 15) * HDIM + kbase + (lane >> 4) * 8;
  v16bf a;
#pragma unroll
  for (int i = 0; i < 8; ++i) { a[i] = p[i]; a[i + 8] = p[16 + i]; }
  return a;
}

// B-fragment (32x16 bf16) for h @ W^T from a row-major bf16 weight (global or
// LDS): lane L holds column n = nbase + L%16; halves 0..7 -> K = kbase +
// (L>>4)*8 + {0..7}, halves 8..15 -> K = kbase + 16 + (L>>4)*8 + {0..7}.
__device__ __forceinline__ v16bf load_b_bf16(const __bf16* __restrict__ W,
                                             int ld, int nbase, int kbase,
                                             int lane) {
  const __bf16* p = W + (nbase + (lane & 15)) * ld + kbase + (lane >> 4) * 8;
  v16bf b;
#pragma unroll
  for (int i = 0; i < 8; ++i) { b[i] = p[i]; b[i + 8] = p[16 + i]; }
  return b;
}

// ---------------------------------------------------------------------------
// Kernel 0: pack all weights to bf16 once:  [W1 | W2 | Wk | Wq | Wv | Ws]
// ---------------------------------------------------------------------------
__global__ __launch_bounds__(256) void pack_weights_kernel(
    const float* __restrict__ W1, const float* __restrict__ W2,
    const float* __restrict__ Wk, const float* __restrict__ Wq,
    const float* __restrict__ Wv, const float* __restrict__ Ws,
    __bf16* __restrict__ out)
{
  int i = blockIdx.x * 256 + threadIdx.x;
  if (i >= PACK_ELEMS) return;
  float v;
  if      (i < W1_ELEMS)                v = W1[i];
  else if (i < W1_ELEMS + 1 * WS_ELEMS) v = W2[i - W1_ELEMS];
  else if (i < W1_ELEMS + 2 * WS_ELEMS) v = Wk[i - W1_ELEMS - 1 * WS_ELEMS];
  else if (i < W1_ELEMS + 3 * WS_ELEMS) v = Wq[i - W1_ELEMS - 2 * WS_ELEMS];
  else if (i < W1_ELEMS + 4 * WS_ELEMS) v = Wv[i - W1_ELEMS - 3 * WS_ELEMS];
  else                                  v = Ws[i - W1_ELEMS - 4 * WS_ELEMS];
  out[i] = (__bf16)v;
}

// ---------------------------------------------------------------------------
// Kernel 1: per 16-node tile (1 wave each, 8 waves/block):
//   h1 = relu(x @ W1^T + b1); h = h1 @ W2^T + b2
//   k = h@Wk^T+bk; q = h@Wq^T+bq; v = h@Wv^T+bv; agg = h@Ws^T + b_gate (seed)
// 64x64 weights staged in LDS (async-to-LDS when available); W1 read as bf16
// straight from the packed global copy.
// ---------------------------------------------------------------------------
__global__ __launch_bounds__(256) void node_dense_kernel(
    const float* __restrict__ x, const __bf16* __restrict__ wpack,
    const float* __restrict__ b1, const float* __restrict__ b2,
    const float* __restrict__ bk, const float* __restrict__ bq,
    const float* __restrict__ bv, const float* __restrict__ bg,
    float* __restrict__ kout, float* __restrict__ qout,
    float* __restrict__ vout, float* __restrict__ agg, int N)
{
  __shared__ __align__(16) __bf16 lds_w[5 * WS_ELEMS];   // W2,Wk,Wq,Wv,Ws (40 KB)
  __shared__ __align__(16) __bf16 lds_h[8][16 * HDIM];   // per-wave h tiles (16 KB)

  const int tid  = threadIdx.x;
  const int lane = tid & 31;
  const int wave = tid >> 5;
  const int kgrp = lane >> 4;              // 0/1: K half select
  const int nlan = lane & 15;              // column within 16-wide tile
  const int node_base = blockIdx.x * 128 + wave * 16;

  // ---- stage the five 64x64 bf16 matrices into LDS (2560 x 16B chunks) -----
  {
    const char* gsrc = (const char*)(wpack + W1_ELEMS);
    char*       ldst = (char*)lds_w;
#if HAVE_ASYNC_LDS
#pragma unroll
    for (int c = 0; c < 10; ++c) {
      int off = (c * 256 + tid) * 16;
      ASYNC_COPY_B128(gsrc + off, ldst + off);
    }
    WAIT_ASYNC();
#else
    const uint4* gs = (const uint4*)gsrc;
    uint4*       ls = (uint4*)ldst;
#pragma unroll
    for (int c = 0; c < 10; ++c) ls[c * 256 + tid] = gs[c * 256 + tid];
#endif
  }
  __syncthreads();

  const __bf16* lW2 = lds_w;
  const __bf16* lWk = lds_w + 1 * WS_ELEMS;
  const __bf16* lWq = lds_w + 2 * WS_ELEMS;
  const __bf16* lWv = lds_w + 3 * WS_ELEMS;
  const __bf16* lWs = lds_w + 4 * WS_ELEMS;

  int rA = node_base + nlan;               // A-operand row, clamped for tail
  if (rA >= N) rA = N - 1;
  const float* xrow = x + (size_t)rA * F_IN;

  const v8f zero8 = {0.f, 0.f, 0.f, 0.f, 0.f, 0.f, 0.f, 0.f};

  // ---- layer 1: K = 128, 4 K-steps x 4 N-tiles = 16 WMMAs -------------------
  v8f acc[4] = {zero8, zero8, zero8, zero8};
#pragma unroll
  for (int kt = 0; kt < 4; ++kt) {
    v16bf a = load_a_f32row(xrow, kt * 32, kgrp);
#pragma unroll
    for (int nt = 0; nt < 4; ++nt)
      acc[nt] = wmma_bf16(a, load_b_bf16(wpack, F_IN, nt * 16, kt * 32, lane),
                          acc[nt]);
  }
  // bias + relu, stage bf16 tile to LDS (C layout: VGPR r -> local row r+8*kgrp)
#pragma unroll
  for (int nt = 0; nt < 4; ++nt) {
    int n = nt * 16 + nlan;
    float bias = b1[n];
#pragma unroll
    for (int r = 0; r < 8; ++r) {
      float val = acc[nt][r] + bias;
      val = val > 0.f ? val : 0.f;
      lds_h[wave][(8 * kgrp + r) * HDIM + n] = (__bf16)val;
    }
  }
  __syncthreads();

  // ---- layer 2: K = 64, 2 K-steps x 4 N-tiles = 8 WMMAs ---------------------
  v8f hacc[4] = {zero8, zero8, zero8, zero8};
#pragma unroll
  for (int kt = 0; kt < 2; ++kt) {
    v16bf a = load_a_lds(lds_h[wave], kt * 32, lane);
#pragma unroll
    for (int nt = 0; nt < 4; ++nt)
      hacc[nt] = wmma_bf16(a, load_b_bf16(lW2, HDIM, nt * 16, kt * 32, lane),
                           hacc[nt]);
  }
  __syncthreads();
#pragma unroll
  for (int nt = 0; nt < 4; ++nt) {
    int n = nt * 16 + nlan;
    float bias = b2[n];
#pragma unroll
    for (int r = 0; r < 8; ++r)
      lds_h[wave][(8 * kgrp + r) * HDIM + n] = (__bf16)(hacc[nt][r] + bias);
  }
  __syncthreads();

  // ---- fused k/q/v/skip projections: 2 K x 4 N x 4 mats = 32 WMMAs ----------
  v8f ka[4] = {zero8, zero8, zero8, zero8};
  v8f qa[4] = {zero8, zero8, zero8, zero8};
  v8f va[4] = {zero8, zero8, zero8, zero8};
  v8f sa[4] = {zero8, zero8, zero8, zero8};
  v16bf a0 = load_a_lds(lds_h[wave], 0, lane);
  v16bf a1 = load_a_lds(lds_h[wave], 32, lane);
#pragma unroll
  for (int kt = 0; kt < 2; ++kt) {
    v16bf a = kt ? a1 : a0;
    int kb = kt * 32;
#pragma unroll
    for (int nt = 0; nt < 4; ++nt) {
      ka[nt] = wmma_bf16(a, load_b_bf16(lWk, HDIM, nt * 16, kb, lane), ka[nt]);
      qa[nt] = wmma_bf16(a, load_b_bf16(lWq, HDIM, nt * 16, kb, lane), qa[nt]);
      va[nt] = wmma_bf16(a, load_b_bf16(lWv, HDIM, nt * 16, kb, lane), va[nt]);
      sa[nt] = wmma_bf16(a, load_b_bf16(lWs, HDIM, nt * 16, kb, lane), sa[nt]);
    }
  }

  // ---- store k/q/v and seed agg with skip + conv bias -----------------------
#pragma unroll
  for (int nt = 0; nt < 4; ++nt) {
    int n = nt * 16 + nlan;
    float bk_ = bk[n], bq_ = bq[n], bv_ = bv[n], bg_ = bg[n];
#pragma unroll
    for (int r = 0; r < 8; ++r) {
      int row = node_base + 8 * kgrp + r;
      if (row < N) {
        size_t off = (size_t)row * HDIM + n;
        kout[off] = ka[nt][r] + bk_;
        qout[off] = qa[nt][r] + bq_;
        vout[off] = va[nt][r] + bv_;
        agg[off]  = sa[nt][r] + bg_;
      }
    }
  }
}

// ---------------------------------------------------------------------------
// Kernel 2: edges. 16 lanes/edge, 4 features/lane.
// msg = sigmoid(k[dst] + q[src]) * v[src]; atomic add into agg[dst].
// k/q/v/agg are 25.6 MB each -> resident in the 192 MB L2.
// ---------------------------------------------------------------------------
__global__ __launch_bounds__(256) void edge_kernel(
    const long long* __restrict__ ei,
    const float* __restrict__ k, const float* __restrict__ q,
    const float* __restrict__ v, float* __restrict__ agg, long long E)
{
  long long t = (long long)blockIdx.x * 256 + threadIdx.x;
  long long e = t >> 4;
  if (e >= E) return;
  int f = (int)(t & 15) * 4;

  long long s = ei[e];        // source j
  long long d = ei[E + e];    // destination i

  const float4 kk = *(const float4*)(k + (size_t)d * HDIM + f);
  const float4 qq = *(const float4*)(q + (size_t)s * HDIM + f);
  const float4 vv = *(const float4*)(v + (size_t)s * HDIM + f);

  float4 m;
  m.x = vv.x / (1.f + __expf(-(kk.x + qq.x)));
  m.y = vv.y / (1.f + __expf(-(kk.y + qq.y)));
  m.z = vv.z / (1.f + __expf(-(kk.z + qq.z)));
  m.w = vv.w / (1.f + __expf(-(kk.w + qq.w)));

  float* dst = agg + (size_t)d * HDIM + f;
  atomicAdd(dst + 0, m.x);
  atomicAdd(dst + 1, m.y);
  atomicAdd(dst + 2, m.z);
  atomicAdd(dst + 3, m.w);
}

// ---------------------------------------------------------------------------
// Kernel 3: scorer. One wave per node, shuffle reduction over 64 features.
// ---------------------------------------------------------------------------
__global__ __launch_bounds__(256) void score_kernel(
    const float* __restrict__ agg, const float* __restrict__ wsc,
    const float* __restrict__ bsc, float* __restrict__ out, int N)
{
  int n = blockIdx.x * 8 + (threadIdx.x >> 5);
  int lane = threadIdx.x & 31;
  if (n >= N) return;
  const float* row = agg + (size_t)n * HDIM;
  float vsum = row[lane] * wsc[lane] + row[32 + lane] * wsc[32 + lane];
#pragma unroll
  for (int off = 16; off > 0; off >>= 1)
    vsum += __shfl_down(vsum, off, 32);
  if (lane == 0) out[n] = vsum + bsc[0];
}

// ---------------------------------------------------------------------------
extern "C" void kernel_launch(void* const* d_in, const int* in_sizes, int n_in,
                              void* d_out, int out_size, void* d_ws, size_t ws_size,
                              hipStream_t stream)
{
  const float*     x   = (const float*)d_in[0];
  const long long* ei  = (const long long*)d_in[1];
  const float* W1  = (const float*)d_in[2];  const float* b1  = (const float*)d_in[3];
  const float* W2  = (const float*)d_in[4];  const float* b2  = (const float*)d_in[5];
  const float* Wk  = (const float*)d_in[6];  const float* bk  = (const float*)d_in[7];
  const float* Wq  = (const float*)d_in[8];  const float* bq  = (const float*)d_in[9];
  const float* Wv  = (const float*)d_in[10]; const float* bv  = (const float*)d_in[11];
  const float* Ws  = (const float*)d_in[12]; const float* bg  = (const float*)d_in[13];
  const float* Wsc = (const float*)d_in[14]; const float* bsc = (const float*)d_in[15];

  const int       N = in_sizes[0] / F_IN;
  const long long E = in_sizes[1] / 2;

  // workspace: k, q, v, agg (each N*64 f32), then bf16 weight pack
  float* kbuf = (float*)d_ws;
  float* qbuf = kbuf + (size_t)N * HDIM;
  float* vbuf = qbuf + (size_t)N * HDIM;
  float* agg  = vbuf + (size_t)N * HDIM;
  __bf16* wpack = (__bf16*)(agg + (size_t)N * HDIM);

  pack_weights_kernel<<<dim3((PACK_ELEMS + 255) / 256), dim3(256), 0, stream>>>(
      W1, W2, Wk, Wq, Wv, Ws, wpack);

  node_dense_kernel<<<dim3((N + 127) / 128), dim3(256), 0, stream>>>(
      x, wpack, b1, b2, bk, bq, bv, bg, kbuf, qbuf, vbuf, agg, N);

  unsigned int eblocks = (unsigned int)((E * 16 + 255) / 256);
  edge_kernel<<<dim3(eblocks), dim3(256), 0, stream>>>(ei, kbuf, qbuf, vbuf, agg, E);

  score_kernel<<<dim3((N + 7) / 8), dim3(256), 0, stream>>>(
      agg, Wsc, bsc, (float*)d_out, N);
}
